// STGCN_8254927143319
// MI455X (gfx1250) — matrix operand order repaired and checked by reading, not compiled, not run on hardware
//
#include <hip/hip_runtime.h>
#include <stdint.h>

// Problem constants (match reference)
#define B_SZ        2048
#define N_NODES     64
#define C_INF       512
#define C_HID       128
#define N_CLS       2
#define E_EDGES     1024
#define EN          (E_EDGES + N_NODES)   // 1088 edges incl. self loops

// LDS row strides (halves), padded to odd dword stride for conflict-free
// wave32 fragment loads.
#define SA_STRIDE   34      // A chunk rows (32 K-halves + pad)
#define SB_STRIDE   34      // B chunk rows (32 K-halves + pad)
#define SA2_STRIDE  130     // hidden bf16 rows (128 + pad)

typedef __attribute__((ext_vector_type(16))) __bf16       v16bf;
typedef __attribute__((ext_vector_type(8)))  float        v8f;
typedef __attribute__((ext_vector_type(8)))  unsigned int v8u;

// Packed fp32x2 -> bf16x2 (hardware round-to-nearest-even), single VALU op.
__device__ __forceinline__ unsigned int pk_bf16(float lo, float hi) {
    unsigned int d;
    asm volatile("v_cvt_pk_bf16_f32 %0, %1, %2" : "=v"(d) : "v"(lo), "v"(hi));
    return d;
}

// A fragment (16x32 bf16), contiguous K order (k = 16*half + s), same as load_b_frag.
// p: K-contiguous row-major bf16 (as ushort), rowstride in elements.
__device__ __forceinline__ v16bf load_a_frag(const unsigned short* p, int rowstride,
                                             int mrow_base, int k0, int lane) {
    const int m  = mrow_base + (lane & 15);
    const int kh = lane >> 4;                 // K-half select
    const unsigned short* rp = p + m * rowstride + k0;
    v8u au;
#pragma unroll
    for (int v = 0; v < 8; ++v) {
        int kb = 16 * kh + 2 * v;
        au[v] = *(const unsigned int*)(rp + kb);   // kb even -> dword aligned
    }
    return __builtin_bit_cast(v16bf, au);
}

// B fragment (32x16 bf16). pn: column-major staging [n][SB_STRIDE] (K contiguous).
__device__ __forceinline__ v16bf load_b_frag(const unsigned short* pn, int n_base, int lane) {
    const int n    = n_base + (lane & 15);
    const int koff = 16 * (lane >> 4);
    const unsigned short* rp = pn + n * SB_STRIDE + koff;
    v8u bu;
#pragma unroll
    for (int v = 0; v < 8; ++v) bu[v] = *(const unsigned int*)(rp + 2 * v);
    return __builtin_bit_cast(v16bf, bu);
}

// ---------------------------------------------------------------------------
// Setup: degree -> rsqrt norm -> per-edge coef -> dst-sorted CSR (atomic-free
// aggregation later). One block, runs once per launch (deterministic).
// ---------------------------------------------------------------------------
__global__ __launch_bounds__(256)
void gcn_setup_kernel(const int* __restrict__ ei, int* __restrict__ rowptr,
                      int* __restrict__ srcs, float* __restrict__ wcoef) {
    __shared__ int   deg[N_NODES];
    __shared__ float nrm[N_NODES];
    __shared__ int   rp[N_NODES + 1];
    __shared__ int   cur[N_NODES];
    const int tid = threadIdx.x;

    if (tid < N_NODES) deg[tid] = 0;
    __syncthreads();
    for (int e = tid; e < EN; e += 256) {
        int dst = (e < E_EDGES) ? (ei[E_EDGES + e] & (N_NODES - 1)) : (e - E_EDGES);
        atomicAdd(&deg[dst], 1);
    }
    __syncthreads();
    if (tid < N_NODES) {
        int d = deg[tid]; if (d < 1) d = 1;
        nrm[tid] = rsqrtf((float)d);
    }
    __syncthreads();
    if (tid == 0) {
        rp[0] = 0;
        for (int n = 0; n < N_NODES; ++n) rp[n + 1] = rp[n] + deg[n];
    }
    __syncthreads();
    if (tid < N_NODES) cur[tid] = rp[tid];
    if (tid <= N_NODES) rowptr[tid] = rp[tid];
    __syncthreads();
    for (int e = tid; e < EN; e += 256) {
        int src = (e < E_EDGES) ? (ei[e] & (N_NODES - 1)) : (e - E_EDGES);
        int dst = (e < E_EDGES) ? (ei[E_EDGES + e] & (N_NODES - 1)) : (e - E_EDGES);
        int slot = atomicAdd(&cur[dst], 1);
        srcs[slot]  = src;
        wcoef[slot] = nrm[src] * nrm[dst];
    }
}

// ---------------------------------------------------------------------------
// Fused per-batch pipeline: GEMM1 -> agg+ReLU -> GEMM2 -> agg+ReLU -> FC.
// 1 block = 1 batch. 8 waves; wave owns a 16x64 strip (4 WMMA tiles).
// ---------------------------------------------------------------------------
__global__ __launch_bounds__(256)
void gcn_fused_kernel(const float* __restrict__ x,  const float* __restrict__ W1,
                      const float* __restrict__ b1, const float* __restrict__ W2,
                      const float* __restrict__ b2, const float* __restrict__ fcW,
                      const float* __restrict__ fcb,
                      const int* __restrict__ g_rowptr, const int* __restrict__ g_srcs,
                      const float* __restrict__ g_wcoef, float* __restrict__ out) {
    __shared__ unsigned short sA[2][N_NODES * SA_STRIDE];   // ~8.5 KB  A chunk (bf16)
    __shared__ unsigned short sB[2][C_HID * SB_STRIDE];     // ~17 KB   B chunk, [n][k]
    __shared__ float          sH[N_NODES * C_HID];          // 32 KB    GEMM out / reduce
    __shared__ unsigned short sA2[N_NODES * SA2_STRIDE];    // ~16.3 KB bf16 hidden
    __shared__ float          sG[N_NODES * C_HID];          // 32 KB    FC input
    __shared__ int            s_rowptr[N_NODES + 1];
    __shared__ int            s_srcs[EN];
    __shared__ float          s_wcoef[EN];

    const int tid  = threadIdx.x;
    const int lane = tid & 31;
    const int wv   = tid >> 5;        // 0..7
    const int mt   = wv >> 1;         // row-tile 0..3 (16 rows each)
    const int ng   = wv & 1;          // col half: n-tiles ng*4 .. ng*4+3
    const int b    = blockIdx.x;
    const float* xb = x + (size_t)b * N_NODES * C_INF;

    // Load CSR of normalized adjacency into LDS
    for (int i = tid; i <= N_NODES; i += 256) s_rowptr[i] = g_rowptr[i];
    for (int i = tid; i < EN; i += 256) { s_srcs[i] = g_srcs[i]; s_wcoef[i] = g_wcoef[i]; }

    v8f acc[4];
#pragma unroll
    for (int j = 0; j < 4; ++j) acc[j] = (v8f){0.f,0.f,0.f,0.f,0.f,0.f,0.f,0.f};

    // --- staging helpers: fp32 global -> packed bf16 LDS dwords ---
    auto stageA1 = [&](int kk, int bsel) {
        const float* src = xb + kk * 32;
#pragma unroll
        for (int r = 0; r < 4; ++r) {               // 1024 pairs
            int p = r * 256 + tid;
            int m = p >> 4, kp = p & 15;            // kp = K-pair index
            float2 f = *(const float2*)(src + m * C_INF + 2 * kp);
            *(unsigned int*)&sA[bsel][m * SA_STRIDE + 2 * kp] = pk_bf16(f.x, f.y);
        }
    };
    auto stageB = [&](const float* W, int kk, int bsel) {
        const float* src = W + kk * 32 * C_HID;
#pragma unroll
        for (int r = 0; r < 8; ++r) {               // 2048 pairs
            int p = r * 256 + tid;
            int n = p & 127, kp = p >> 7;           // n fastest -> coalesced reads
            float f0 = src[(2 * kp)     * C_HID + n];
            float f1 = src[(2 * kp + 1) * C_HID + n];
            *(unsigned int*)&sB[bsel][n * SB_STRIDE + 2 * kp] = pk_bf16(f0, f1);
        }
    };
    auto store_acc = [&]() {
        const int colb = lane & 15;
        const int r0   = mt * 16 + 8 * (lane >> 4);
#pragma unroll
        for (int j = 0; j < 4; ++j) {
            int nc = (ng * 4 + j) * 16 + colb;
#pragma unroll
            for (int r = 0; r < 8; ++r)
                sH[(r0 + r) * C_HID + nc] = acc[j][r];
        }
    };
    // CSR aggregate + bias + ReLU over 2 channels per thread.
    auto aggregate = [&](const float* bias, bool to_bf16) {
#pragma unroll 2
        for (int it = 0; it < 16; ++it) {           // 4096 channel-pairs
            int p = it * 256 + tid;
            int n = p >> 6, c = (p & 63) * 2;
            float s0 = 0.f, s1 = 0.f;
            int e1 = s_rowptr[n + 1];
            for (int e = s_rowptr[n]; e < e1; ++e) {
                float  w = s_wcoef[e];
                float2 h = *(const float2*)(sH + s_srcs[e] * C_HID + c);
                s0 = fmaf(h.x, w, s0);
                s1 = fmaf(h.y, w, s1);
            }
            float2 bb = *(const float2*)(bias + c);
            s0 = fmaxf(s0 + bb.x, 0.f);
            s1 = fmaxf(s1 + bb.y, 0.f);
            if (to_bf16) {
                *(unsigned int*)&sA2[n * SA2_STRIDE + c] = pk_bf16(s0, s1);
            } else {
                *(float2*)(sG + n * C_HID + c) = make_float2(s0, s1);
            }
        }
    };

    // ================= GEMM1: [64,512]x[512,128], 16 K-steps =================
    stageA1(0, 0); stageB(W1, 0, 0);
    __syncthreads();
    int buf = 0;
    for (int kk = 0; kk < C_INF / 32; ++kk) {
        if (kk + 1 < C_INF / 32) { stageA1(kk + 1, buf ^ 1); stageB(W1, kk + 1, buf ^ 1); }
        v16bf a = load_a_frag(sA[buf], SA_STRIDE, mt * 16, 0, lane);
#pragma unroll
        for (int j = 0; j < 4; ++j) {
            v16bf bb = load_b_frag(sB[buf], (ng * 4 + j) * 16, lane);
            acc[j] = __builtin_amdgcn_wmma_f32_16x16x32_bf16(
                false, a, false, bb, (short)0, acc[j], false, false);
        }
        __syncthreads();
        buf ^= 1;
    }
    store_acc();
    __syncthreads();

    // agg1 + bias + ReLU -> bf16 A for GEMM2; overlap with staging W2 chunk 0
    aggregate(b1, true);
    stageB(W2, 0, 0);
    __syncthreads();

    // ================= GEMM2: [64,128]x[128,128], 4 K-steps ==================
#pragma unroll
    for (int j = 0; j < 4; ++j) acc[j] = (v8f){0.f,0.f,0.f,0.f,0.f,0.f,0.f,0.f};
    buf = 0;
    for (int kk = 0; kk < C_HID / 32; ++kk) {
        if (kk + 1 < C_HID / 32) stageB(W2, kk + 1, buf ^ 1);
        v16bf a = load_a_frag(sA2, SA2_STRIDE, mt * 16, kk * 32, lane);
#pragma unroll
        for (int j = 0; j < 4; ++j) {
            v16bf bb = load_b_frag(sB[buf], (ng * 4 + j) * 16, lane);
            acc[j] = __builtin_amdgcn_wmma_f32_16x16x32_bf16(
                false, a, false, bb, (short)0, acc[j], false, false);
        }
        __syncthreads();
        buf ^= 1;
    }
    store_acc();
    __syncthreads();

    // agg2 + bias + ReLU -> sG (fp32, the flattened [64*128] FC input)
    aggregate(b2, false);
    __syncthreads();

    // ================= FC: [8192] . fcW[8192,2] + fcb =======================
    float p0 = 0.f, p1 = 0.f;
    for (int i = tid; i < N_NODES * C_HID; i += 256) {
        float v = sG[i];
        float2 w = *(const float2*)(fcW + 2 * i);
        p0 = fmaf(v, w.x, p0);
        p1 = fmaf(v, w.y, p1);
    }
    float* sred = sH;                 // sH free now; reuse for reduction
    sred[tid]       = p0;
    sred[256 + tid] = p1;
    __syncthreads();
    for (int s = 128; s > 0; s >>= 1) {
        if (tid < s) {
            sred[tid]       += sred[tid + s];
            sred[256 + tid] += sred[256 + tid + s];
        }
        __syncthreads();
    }
    if (tid == 0) {
        out[(size_t)b * N_CLS + 0] = sred[0]   + fcb[0];
        out[(size_t)b * N_CLS + 1] = sred[256] + fcb[1];
    }
}

// ---------------------------------------------------------------------------
extern "C" void kernel_launch(void* const* d_in, const int* in_sizes, int n_in,
                              void* d_out, int out_size, void* d_ws, size_t ws_size,
                              hipStream_t stream) {
    const float* x   = (const float*)d_in[0];
    const int*   ei  = (const int*)  d_in[1];
    const float* W1  = (const float*)d_in[2];
    const float* b1  = (const float*)d_in[3];
    const float* W2  = (const float*)d_in[4];
    const float* b2  = (const float*)d_in[5];
    const float* fcW = (const float*)d_in[6];
    const float* fcb = (const float*)d_in[7];
    float* out = (float*)d_out;

    // workspace: rowptr[65] | srcs[1088] | wcoef[1088]
    int*   ws_rowptr = (int*)d_ws;
    int*   ws_srcs   = ws_rowptr + 128;
    float* ws_wcoef  = (float*)(ws_srcs + EN + 32);

    gcn_setup_kernel<<<1, 256, 0, stream>>>(ei, ws_rowptr, ws_srcs, ws_wcoef);
    gcn_fused_kernel<<<B_SZ, 256, 0, stream>>>(x, W1, b1, W2, b2, fcW, fcb,
                                               ws_rowptr, ws_srcs, ws_wcoef, out);
}
